// HydraNeighborhoodAttention_66219805769959
// MI455X (gfx1250) — matrix operand order repaired and checked by reading, not compiled, not run on hardware
//
#include <hip/hip_runtime.h>
#include <hip/hip_bf16.h>

typedef __attribute__((ext_vector_type(16))) __bf16 v16bf;
typedef __attribute__((ext_vector_type(8)))  float  v8f;
typedef unsigned int u32x4 __attribute__((ext_vector_type(4)));
typedef unsigned int u32x8 __attribute__((ext_vector_type(8)));
typedef int          i32x4 __attribute__((ext_vector_type(4)));
typedef int          i32x8 __attribute__((ext_vector_type(8)));

#if defined(__has_builtin)
#  if __has_builtin(__builtin_amdgcn_tensor_load_to_lds) && __has_builtin(__builtin_amdgcn_s_wait_tensorcnt)
#    define HAS_TDM 1
#  else
#    define HAS_TDM 0
#  endif
#else
#  define HAS_TDM 0
#endif

union FragBF16 {
    uint4 u[2];
    v16bf v;
};

__device__ __forceinline__ unsigned short f32_to_bf16(float f) {
    unsigned int u = __float_as_uint(f);
    if ((u & 0x7fffffffu) > 0x7f800000u)            // NaN: keep quiet NaN
        return (unsigned short)((u >> 16) | 0x40u);
    u += 0x7fffu + ((u >> 16) & 1u);                // round to nearest even
    return (unsigned short)(u >> 16);
}

#if HAS_TDM
// ---------------------------------------------------------------------------
// Issue a TDM 2D tile load: global (row-major, bf16, row stride stride_x elems)
// -> LDS at byte offset lds_off, tile_x (contiguous) x tile_y rows.
// D# fields per CDNA5 ISA 8.3/8.4: group0 = {count/type/lds/global},
// group1 = {data_size=2B, tensor dims, tile dims, dim0 stride}.
// This toolchain's builtin takes 6 args (g0, g1, g2, g3, g4, cpol); the extra
// descriptor groups are zero (2D tensor, not in a cluster).
// ---------------------------------------------------------------------------
__device__ __forceinline__ void tdm_load_2d(unsigned lds_off, const void* gptr,
                                            unsigned tile_x, unsigned tile_y,
                                            unsigned tensor_x, unsigned tensor_y,
                                            unsigned stride_x)
{
    unsigned long long ga = (unsigned long long)(size_t)gptr;
    u32x4 g0;
    g0[0] = 1u;                                           // count=1, is_restore=0
    g0[1] = lds_off;                                      // lds_addr (bytes)
    g0[2] = (unsigned)(ga & 0xffffffffu);                 // global_addr[31:0]
    g0[3] = (unsigned)((ga >> 32) & 0x01ffffffu)          // global_addr[56:32]
          | (2u << 30);                                   // type = 2 ("image")
    u32x8 g1u;
    g1u[0] = (1u << 16);                                  // data_size=1 (2 bytes)
    g1u[1] = (tensor_x & 0xffffu) << 16;                  // tensor_dim0[15:0]
    g1u[2] = ((tensor_x >> 16) & 0xffffu)                 // tensor_dim0[31:16]
           | ((tensor_y & 0xffffu) << 16);                // tensor_dim1[15:0]
    g1u[3] = ((tensor_y >> 16) & 0xffffu)                 // tensor_dim1[31:16]
           | (tile_x << 16);                              // tile_dim0
    g1u[4] = tile_y;                                      // tile_dim1 (tile_dim2=0)
    g1u[5] = stride_x;                                    // tensor_dim0_stride[31:0]
    g1u[6] = 0;
    g1u[7] = 0;
    i32x8 g1 = __builtin_bit_cast(i32x8, g1u);
    i32x4 z4 = {0, 0, 0, 0};
    i32x8 z8 = {0, 0, 0, 0, 0, 0, 0, 0};
    __builtin_amdgcn_tensor_load_to_lds(g0, g1, z4, z4, z8, 0);
}
#endif

// ---------------------------------------------------------------------------
// fp32 -> bf16 elementwise conversion
// ---------------------------------------------------------------------------
__global__ void f32_to_bf16_kernel(const float* __restrict__ src,
                                   unsigned short* __restrict__ dst, int n) {
    int i = blockIdx.x * blockDim.x + threadIdx.x;
    if (i < n) dst[i] = f32_to_bf16(src[i]);
}

// ---------------------------------------------------------------------------
// bf16 GEMM with v_wmma_f32_16x16x32_bf16, fp32 accumulate.
//   A: [M][K] bf16 row-major (shared across batch)
//   B: [batch][K][N] bf16
//   C: [batch][M][N] fp32
// Block = 128 threads (4 waves), 64x64 tile, K-step 32, double-buffered LDS.
// A tile staged by the Tensor Data Mover (TDM) when available; B tile staged
// through registers with an on-the-fly [k][n]->[n][k] transpose so WMMA
// fragments are two contiguous ds_load_b128 each.
// M, N, K must be multiples of 64/64/32 (true for all uses here).
// ---------------------------------------------------------------------------
__global__ void __launch_bounds__(128)
gemm_bf16_wmma(const unsigned short* __restrict__ A,
               const unsigned short* __restrict__ Bm,
               float* __restrict__ Cm,
               int M, int N, int K)
{
    __shared__ __align__(16) unsigned short As[2][64 * 32];   // [m][k]
    __shared__ __align__(16) unsigned short Bs[2][64 * 32];   // [n][k]

    const int t    = threadIdx.x;
    const int lane = t & 31;
    const int wave = t >> 5;
    const int hs   = lane >> 4;     // half-select (lanes 0-15 vs 16-31)
    const int l15  = lane & 15;

    const int n0 = blockIdx.x * 64;
    const int m0 = blockIdx.y * 64;
    const int b  = blockIdx.z;

    const unsigned short* B = Bm + (size_t)b * K * N;
    float* C = Cm + (size_t)b * M * N;

    // per-thread B staging coordinates: 4 units of 4 contiguous n elements
    int krow[4], nc[4];
    #pragma unroll
    for (int it = 0; it < 4; ++it) {
        int u = it * 128 + t;
        krow[it] = u >> 4;          // 0..31
        nc[it]   = (u & 15) * 4;    // 0..60
    }

    uint2 breg[4];
#if !HAS_TDM
    uint4 areg[2];
#endif

    auto load_gB = [&](int k0) {
        #pragma unroll
        for (int it = 0; it < 4; ++it)
            breg[it] = *(const uint2*)(&B[(size_t)(k0 + krow[it]) * N + n0 + nc[it]]);
    };
    auto store_sB = [&](int buf) {
        unsigned short* bs = &Bs[buf][0];
        #pragma unroll
        for (int it = 0; it < 4; ++it) {
            bs[(nc[it] + 0) * 32 + krow[it]] = (unsigned short)(breg[it].x & 0xffffu);
            bs[(nc[it] + 1) * 32 + krow[it]] = (unsigned short)(breg[it].x >> 16);
            bs[(nc[it] + 2) * 32 + krow[it]] = (unsigned short)(breg[it].y & 0xffffu);
            bs[(nc[it] + 3) * 32 + krow[it]] = (unsigned short)(breg[it].y >> 16);
        }
    };
#if HAS_TDM
    auto issue_A = [&](int k0, int buf) {
        if (wave == 0)
            tdm_load_2d((unsigned)(size_t)(void*)&As[buf][0],
                        A + (size_t)m0 * K + k0,
                        32u, 64u, (unsigned)K, (unsigned)M, (unsigned)K);
    };
#else
    auto load_gA = [&](int k0) {
        #pragma unroll
        for (int c = 0; c < 2; ++c) {
            int cc = c * 128 + t, row = cc >> 2, col = cc & 3;
            areg[c] = *(const uint4*)(&A[(size_t)(m0 + row) * K + k0 + col * 8]);
        }
    };
    auto store_sA = [&](int buf) {
        #pragma unroll
        for (int c = 0; c < 2; ++c) {
            int cc = c * 128 + t, row = cc >> 2, col = cc & 3;
            *(uint4*)(&As[buf][row * 32 + col * 8]) = areg[c];
        }
    };
#endif

    v8f acc[4];
    #pragma unroll
    for (int i = 0; i < 4; ++i) acc[i] = (v8f){};

    // ---- prologue: stage k-tile 0 into buffer 0 ----
#if HAS_TDM
    issue_A(0, 0);
#else
    load_gA(0);
    store_sA(0);
#endif
    load_gB(0);
    store_sB(0);
#if HAS_TDM
    if (wave == 0) __builtin_amdgcn_s_wait_tensorcnt(0);
#endif
    __syncthreads();

    const int steps = K >> 5;
    for (int kt = 0; kt < steps; ++kt) {
        const int cur = kt & 1, nxt = cur ^ 1;
        const bool more = (kt + 1) < steps;
        const int k0n = (kt + 1) << 5;

        // kick off next tile's data movement (TDM + global loads) before compute
        if (more) {
#if HAS_TDM
            issue_A(k0n, nxt);
#else
            load_gA(k0n);
#endif
            load_gB(k0n);
        }

        // ---- compute current tile ----
        FragBF16 fa;
        {
            const unsigned short* as = &As[cur][0];
            int arow = wave * 16 + l15;
            fa.u[0] = *(const uint4*)(&as[arow * 32 + hs * 8]);       // K = hs*8 + 0..7
            fa.u[1] = *(const uint4*)(&as[arow * 32 + 16 + hs * 8]);  // K = 16+hs*8 + 0..7
        }
        {
            const unsigned short* bsc = &Bs[cur][0];
            #pragma unroll
            for (int nt = 0; nt < 4; ++nt) {
                FragBF16 fb;
                int brow = nt * 16 + l15;                             // N = l15
                fb.u[0] = *(const uint4*)(&bsc[brow * 32 + hs * 8]);      // K = hs*8 + 0..7
                fb.u[1] = *(const uint4*)(&bsc[brow * 32 + 16 + hs * 8]); // K = 16+hs*8 + 0..7
                acc[nt] = __builtin_amdgcn_wmma_f32_16x16x32_bf16(
                    false, fa.v, false, fb.v, (short)0, acc[nt], false, false);
            }
        }

        // ---- finish staging next tile, then one barrier ----
        if (more) {
#if !HAS_TDM
            store_sA(nxt);
#endif
            store_sB(nxt);
#if HAS_TDM
            if (wave == 0) __builtin_amdgcn_s_wait_tensorcnt(0);
#endif
        }
        __syncthreads();
    }

    // ---- epilogue: C/D layout: VGPR r -> M = hs*8 + r, N = l15 ----
    #pragma unroll
    for (int nt = 0; nt < 4; ++nt) {
        #pragma unroll
        for (int r = 0; r < 8; ++r) {
            int m = m0 + wave * 16 + hs * 8 + r;
            int n = n0 + nt * 16 + l15;
            C[(size_t)m * N + n] = acc[nt][r];
        }
    }
}

// ---------------------------------------------------------------------------
// Depthwise 3x3 conv, SAME padding, no bias. in/out: [B][C3][H][W] fp32.
// ---------------------------------------------------------------------------
__global__ void dwconv3x3_kernel(const float* __restrict__ in,
                                 const float* __restrict__ w,
                                 float* __restrict__ out,
                                 int total, int C3, int H, int W)
{
    int idx = blockIdx.x * blockDim.x + threadIdx.x;
    if (idx >= total) return;
    int wc0 = idx % W;
    int tmp = idx / W;
    int hr0 = tmp % H;
    tmp /= H;
    int ch = tmp % C3;
    int b  = tmp / C3;

    const float* inc = in + ((size_t)b * C3 + ch) * H * W;
    const float* wk  = w + ch * 9;
    float s = 0.f;
    #pragma unroll
    for (int r = 0; r < 3; ++r) {
        int hr = hr0 + r - 1;
        if (hr < 0 || hr >= H) continue;
        #pragma unroll
        for (int c = 0; c < 3; ++c) {
            int wc = wc0 + c - 1;
            if (wc < 0 || wc >= W) continue;
            s += inc[hr * W + wc] * wk[r * 3 + c];
        }
    }
    out[idx] = s;
}

// ---------------------------------------------------------------------------
// NATTEN window helpers (match reference _window_start / _pb_start)
// ---------------------------------------------------------------------------
__device__ __forceinline__ int window_start(int idx, int L, int K, int D) {
    int NH = K / 2;
    if (D <= 1) {
        int s = max(idx - NH, 0);
        if (idx + NH >= L) s += (L - idx - NH - 1);
        return s;
    }
    int ni = idx - NH * D;
    int imodd = idx % D;
    int a = (L / D) * D;
    int bb = L - a;
    int edge = (imodd < bb) ? (L - bb + imodd - 2 * NH * D) : (a + imodd - K * D);
    if (ni < 0) return imodd;
    if (idx + NH * D >= L) return edge;
    return ni;
}

__device__ __forceinline__ int pb_start(int idx, int L, int K, int D) {
    int NH = K / 2;
    if (D <= 1) {
        int s = NH;
        if (idx < NH) s += (NH - idx);
        if (idx + NH >= L) s += (L - idx - 1 - NH);
        return s;
    }
    if (idx - NH * D < 0) return K - 1 - idx / D;
    if (idx + NH * D >= L) return (L - idx - 1) / D;
    return NH;
}

// ---------------------------------------------------------------------------
// Neighborhood attention. One thread per query pixel per head per batch.
// z: conv output [B][768][3136] fp32, read with the torch-reshape scramble:
//    q(p,n,d) = z_flat[p*768 + 0   + n*32 + d]
//    k(p,n,d) = z_flat[p*768 + 256 + n*32 + d]
//    v(p,n,d) = z_flat[p*768 + 512 + n*32 + d]
// Output written bf16 into obuf[b][(n*32+d)*HW + p] ([C][HW] layout, ready
// for the proj GEMM).
// ---------------------------------------------------------------------------
__global__ void __launch_bounds__(128)
natten_kernel(const float* __restrict__ z,
              const float* __restrict__ rpb0,
              const float* __restrict__ rpb1,
              unsigned short* __restrict__ obuf,
              int H, int W)
{
    const int HW = H * W;
    const int p = blockIdx.x * blockDim.x + threadIdx.x;
    if (p >= HW) return;
    const int n = blockIdx.y;     // global head 0..7
    const int b = blockIdx.z;

    const int split = n >> 2;
    const int D = split ? 2 : 1;
    const int KS = 7;
    const float* rpb = (split ? rpb1 : rpb0) + (n & 3) * 13 * 13;
    const float* zb = z + (size_t)b * 768 * HW;

    const int i = p / W, j = p % W;
    const int si = window_start(i, H, KS, D);
    const int sj = window_start(j, W, KS, D);
    const int pi = pb_start(i, H, KS, D);
    const int pj = pb_start(j, W, KS, D);

    const float SCALE = 0.17677669529663687f;   // 32^-0.5
    float q[32];
    const float* qp = zb + (size_t)p * 768 + n * 32;
    #pragma unroll
    for (int d = 0; d < 32; ++d) q[d] = qp[d] * SCALE;

    float logits[49];
    float mx = -1e30f;
    #pragma unroll
    for (int u = 0; u < 7; ++u) {
        const int ki = si + u * D;
        #pragma unroll
        for (int vv = 0; vv < 7; ++vv) {
            const int kj = sj + vv * D;
            const float* kp = zb + (size_t)(ki * W + kj) * 768 + 256 + n * 32;
            float s = 0.f;
            #pragma unroll
            for (int d = 0; d < 32; ++d) s += q[d] * kp[d];
            s += rpb[(pi + u) * 13 + (pj + vv)];
            logits[u * 7 + vv] = s;
            mx = fmaxf(mx, s);
        }
    }
    float sum = 0.f;
    #pragma unroll
    for (int e = 0; e < 49; ++e) {
        float w = __expf(logits[e] - mx);
        logits[e] = w;
        sum += w;
    }
    const float inv = 1.f / sum;

    float acc[32];
    #pragma unroll
    for (int d = 0; d < 32; ++d) acc[d] = 0.f;
    #pragma unroll
    for (int u = 0; u < 7; ++u) {
        const int ki = si + u * D;
        #pragma unroll
        for (int vv = 0; vv < 7; ++vv) {
            const int kj = sj + vv * D;
            const float* vp = zb + (size_t)(ki * W + kj) * 768 + 512 + n * 32;
            const float w = logits[u * 7 + vv] * inv;
            #pragma unroll
            for (int d = 0; d < 32; ++d) acc[d] += w * vp[d];
        }
    }

    unsigned short* ob = obuf + (size_t)b * 256 * HW;
    #pragma unroll
    for (int d = 0; d < 32; ++d)
        ob[(size_t)(n * 32 + d) * HW + p] = f32_to_bf16(acc[d]);
}

// ---------------------------------------------------------------------------
// Host orchestration
// ---------------------------------------------------------------------------
extern "C" void kernel_launch(void* const* d_in, const int* in_sizes, int n_in,
                              void* d_out, int out_size, void* d_ws, size_t ws_size,
                              hipStream_t stream)
{
    (void)in_sizes; (void)n_in; (void)out_size; (void)ws_size;

    const float* x      = (const float*)d_in[0];   // [2,256,56,56]
    const float* w_qkv1 = (const float*)d_in[1];   // [768,256]
    const float* w_qkv2 = (const float*)d_in[2];   // [768,1,3,3]
    const float* rpb0   = (const float*)d_in[3];   // [4,13,13]
    const float* rpb1   = (const float*)d_in[4];   // [4,13,13]
    const float* w_proj = (const float*)d_in[5];   // [256,256]
    float* out = (float*)d_out;                    // [2,256,56,56]

    const int B = 2, C = 256, C3 = 768, H = 56, W = 56, HW = H * W;

    // workspace layout
    char* ws = (char*)d_ws;
    size_t off = 0;
    auto wsalloc = [&](size_t bytes) -> void* {
        void* pp = ws + off;
        off = (off + bytes + 255) & ~(size_t)255;
        return pp;
    };
    unsigned short* w1b = (unsigned short*)wsalloc((size_t)C3 * C * 2);       // bf16 w_qkv1
    unsigned short* wpb = (unsigned short*)wsalloc((size_t)C * C * 2);        // bf16 w_proj
    unsigned short* xb  = (unsigned short*)wsalloc((size_t)B * C * HW * 2);   // bf16 x
    float* z1 = (float*)wsalloc((size_t)B * C3 * HW * 4);                     // qkv GEMM out
    float* z2 = (float*)wsalloc((size_t)B * C3 * HW * 4);                     // dw-conv out
    unsigned short* ob = (unsigned short*)wsalloc((size_t)B * C * HW * 2);    // bf16 attn out

    // 1) fp32 -> bf16 conversions
    {
        int n1 = C3 * C;
        f32_to_bf16_kernel<<<(n1 + 255) / 256, 256, 0, stream>>>(w_qkv1, w1b, n1);
        int n2 = C * C;
        f32_to_bf16_kernel<<<(n2 + 255) / 256, 256, 0, stream>>>(w_proj, wpb, n2);
        int n3 = B * C * HW;
        f32_to_bf16_kernel<<<(n3 + 255) / 256, 256, 0, stream>>>(x, xb, n3);
    }

    // 2) qkv 1x1-conv GEMM: [768,256] x [256,3136] per batch (WMMA bf16 + TDM)
    {
        dim3 grid(HW / 64, C3 / 64, B);
        gemm_bf16_wmma<<<grid, 128, 0, stream>>>(w1b, xb, z1, C3, HW, C);
    }

    // 3) depthwise 3x3 conv
    {
        int tot = B * C3 * HW;
        dwconv3x3_kernel<<<(tot + 255) / 256, 256, 0, stream>>>(z1, w_qkv2, z2, tot, C3, H, W);
    }

    // 4) neighborhood attention (both splits; split chosen by head index)
    {
        dim3 grid((HW + 127) / 128, 8, B);
        natten_kernel<<<grid, 128, 0, stream>>>(z2, rpb0, rpb1, ob, H, W);
    }

    // 5) output projection GEMM: [256,256] x [256,3136] per batch (WMMA bf16 + TDM)
    {
        dim3 grid(HW / 64, C / 64, B);
        gemm_bf16_wmma<<<grid, 128, 0, stream>>>(wpb, ob, out, C, HW, C);
    }
}
